// AdaptedMambaBlock_23957327577931
// MI455X (gfx1250) — compile-verified, matrix operand order checked by reading, not run on hardware
//
#include <hip/hip_runtime.h>

#define D_MODEL 1024
#define D_INNER 2048
#define D_STATE 16
#define D_CONV  4
#define DT_RANK 64
#define RANK    16
#define LORA_SCALING 2.0f
#define BATCH   2
#define SEQLEN  2048
#define MROWS   (BATCH * SEQLEN)        // 4096
#define KP_IN   (D_MODEL + 32)          // 1056 (1024 + 16 lora + 16 pad)
#define KP_OUT  (D_INNER + 32)          // 2080
#define NXP     (DT_RANK + 2 * D_STATE) // 96

typedef __attribute__((ext_vector_type(16))) __bf16 v16bf;
typedef __attribute__((ext_vector_type(8)))  float  v8f;

static __device__ __forceinline__ unsigned short f2bf(float f) {
  unsigned int u = __float_as_uint(f);
  u += 0x7FFFu + ((u >> 16) & 1u);   // round-to-nearest-even
  return (unsigned short)(u >> 16);
}

// Build a 16-element bf16 fragment from two contiguous 8-element (16B) chunks.
// Matches the CDNA5 16-bit A/B layout: low lanes hold K=[0..7]+[16..23],
// high lanes hold K=[8..15]+[24..31] of each 32-wide K step.
static __device__ __forceinline__ v16bf load_frag(const unsigned short* p0,
                                                  const unsigned short* p1) {
  uint4 t[2];
  t[0] = *(const uint4*)p0;
  t[1] = *(const uint4*)p1;
  v16bf v;
  __builtin_memcpy(&v, t, sizeof(v));
  return v;
}

static __device__ __forceinline__ float softplusf(float v) {
  return (v > 20.f) ? v : log1pf(__expf(v));
}

// C[M,N] (f32) = A[M,Kp] (bf16, row-major) x W[N,Kp]^T (bf16, row-major).
// Wave computes a 32x64 register tile: 2 A frags x 4 B frags -> 8 WMMAs per
// 6 fragment loads (0.75 loads/WMMA). Block = 8 waves (4M x 2N) = 128x128.
// mode 1: softplus(v + bias[col]) epilogue (dt_proj).
__global__ __launch_bounds__(256)
void gemm_bf16_wmma(const unsigned short* __restrict__ A,
                    const unsigned short* __restrict__ W,
                    float* __restrict__ C,
                    const float* __restrict__ bias,
                    int M, int N, int Kp, int mode)
{
  const int wave = threadIdx.x >> 5;
  const int lane = threadIdx.x & 31;
  const int half = lane >> 4;
  const int lrow = lane & 15;
  const int m0 = blockIdx.y * 128 + (wave & 3) * 32;
  const int n0 = blockIdx.x * 128 + (wave >> 2) * 64;
  if (m0 >= M || n0 >= N) return;
  const int c0 = half ? 8 : 0;
  const int c1 = half ? 24 : 16;

  const unsigned short* arow0 = A + (size_t)(m0 + lrow) * Kp;
  const unsigned short* arow1 = arow0 + (size_t)16 * Kp;
  const unsigned short* brow[4];
  bool bv[4];
#pragma unroll
  for (int j = 0; j < 4; ++j) {
    const int nt = n0 + 16 * j;
    bv[j] = nt < N;
    brow[j] = W + (size_t)((bv[j] ? nt : n0) + lrow) * Kp;
  }

  v8f acc[8];
#pragma unroll
  for (int i = 0; i < 8; ++i) acc[i] = (v8f){};

  for (int k0 = 0; k0 < Kp; k0 += 32) {
    __builtin_prefetch((const void*)(arow0 + k0 + 128), 0, 1);
    __builtin_prefetch((const void*)(brow[0] + k0 + 128), 0, 1);
    const v16bf a0 = load_frag(arow0 + k0 + c0, arow0 + k0 + c1);
    const v16bf a1 = load_frag(arow1 + k0 + c0, arow1 + k0 + c1);
#pragma unroll
    for (int j = 0; j < 4; ++j) {
      const v16bf bf = load_frag(brow[j] + k0 + c0, brow[j] + k0 + c1);
      acc[j]     = __builtin_amdgcn_wmma_f32_16x16x32_bf16(false, a0, false, bf,
                                                           (short)0, acc[j], false, false);
      acc[4 + j] = __builtin_amdgcn_wmma_f32_16x16x32_bf16(false, a1, false, bf,
                                                           (short)0, acc[4 + j], false, false);
    }
  }

  // C layout: lane = column, accum vgpr r -> row (half*8 + r)
#pragma unroll
  for (int j = 0; j < 4; ++j) {
    if (!bv[j]) continue;
    const int col = n0 + 16 * j + lrow;
    const float bcol = (mode == 1) ? bias[col] : 0.f;
#pragma unroll
    for (int r = 0; r < 8; ++r) {
      const int row0 = m0 + half * 8 + r;
      float v = acc[j][r];
      if (mode == 1) v = softplusf(v + bcol);
      C[(size_t)row0 * N + col] = v;
      float w = acc[4 + j][r];
      if (mode == 1) w = softplusf(w + bcol);
      C[(size_t)(row0 + 16) * N + col] = w;
    }
  }
}

// Per row m: bf16-convert X[m,:], append t = X[m,:] @ loraA^T (R values),
// zero-pad to Kp. One block per row.
__global__ __launch_bounds__(256)
void pack_act(const float* __restrict__ X, const float* __restrict__ loraA,
              unsigned short* __restrict__ Ab, int K, int R, int Kp)
{
  __shared__ float red[256];
  const int m = blockIdx.x;
  const int tid = threadIdx.x;
  const float* xr = X + (size_t)m * K;
  unsigned short* ar = Ab + (size_t)m * Kp;
  for (int k = tid; k < K; k += 256) ar[k] = f2bf(xr[k]);
  if (R > 0) {
    const int r = tid >> 4;
    const int sub = tid & 15;
    float p = 0.f;
    const float* a = loraA + (size_t)r * K;
    for (int k = sub; k < K; k += 16) p += xr[k] * a[k];
    red[tid] = p;
    __syncthreads();
    if (sub == 0 && r < R) {
      float s = 0.f;
#pragma unroll
      for (int i = 0; i < 16; ++i) s += red[(r << 4) + i];
      ar[K + r] = f2bf(s);
    }
  }
  for (int k = K + R + tid; k < Kp; k += 256) ar[k] = 0;
}

// Wb[n, :] = [bf16(W[n,:]) | scaling*mask[n]*loraB[n,:] | 0-pad]
__global__ void pack_weight(const float* __restrict__ W, const float* __restrict__ loraB,
                            const float* __restrict__ mask, unsigned short* __restrict__ Wb,
                            int N, int K, int R, int Kp, float scaling)
{
  size_t i = (size_t)blockIdx.x * blockDim.x + threadIdx.x;
  if (i >= (size_t)N * Kp) return;
  const int n = (int)(i / Kp);
  const int k = (int)(i % Kp);
  float v = 0.f;
  if (k < K)          v = W[(size_t)n * K + k];
  else if (k < K + R) v = scaling * mask[n] * loraB[(size_t)n * R + (k - K)];
  Wb[i] = f2bf(v);
}

// Depthwise causal conv1d + SiLU over the first half of xz rows.
__global__ void conv_silu(const float* __restrict__ xz, const float* __restrict__ cw,
                          const float* __restrict__ cb, float* __restrict__ xs,
                          unsigned short* __restrict__ xs_bf)
{
  const int i = blockIdx.x * blockDim.x + threadIdx.x;
  if (i >= MROWS * D_INNER) return;
  const int d = i % D_INNER;
  const int m = i / D_INNER;
  const int l = m % SEQLEN;
  const int b = m / SEQLEN;
  float acc = cb[d];
#pragma unroll
  for (int j = 0; j < D_CONV; ++j) {
    const int ls = l - (D_CONV - 1) + j;
    if (ls >= 0)
      acc += cw[d * D_CONV + j] * xz[(size_t)(b * SEQLEN + ls) * (2 * D_INNER) + d];
  }
  const float s = acc / (1.f + __expf(-acc));
  xs[i] = s;
  xs_bf[i] = f2bf(s);
}

__global__ void extract_dt(const float* __restrict__ xdbl, unsigned short* __restrict__ dt_bf)
{
  const int i = blockIdx.x * blockDim.x + threadIdx.x;
  if (i >= MROWS * DT_RANK) return;
  const int m = i >> 6;
  const int j = i & 63;
  dt_bf[i] = f2bf(xdbl[(size_t)m * NXP + j]);
}

// Selective scan, lane-per-(b,d,n); y reduced across 16-lane groups (wave32
// shuffles). Fused with the SiLU(z) gate. Critical path per step: one FMA.
__global__ __launch_bounds__(256)
void selective_scan(const float* __restrict__ delta, const float* __restrict__ xs,
                    const float* __restrict__ xdbl, const float* __restrict__ A_log,
                    const float* __restrict__ Dp, const float* __restrict__ xz,
                    float* __restrict__ y)
{
  const int g = blockIdx.x * blockDim.x + threadIdx.x;
  if (g >= BATCH * D_INNER * D_STATE) return;
  const int n = g & (D_STATE - 1);
  const int t = g >> 4;
  const int d = t % D_INNER;
  const int b = t / D_INNER;
  const float a  = -__expf(A_log[d * D_STATE + n]);
  const float Dd = Dp[d];
  float h = 0.f;
  for (int l = 0; l < SEQLEN; ++l) {
    const size_t m = (size_t)b * SEQLEN + l;
    const float dlt = delta[m * D_INNER + d];
    const float u   = xs[m * D_INNER + d];
    const float* bc = xdbl + m * NXP + DT_RANK;  // [B(16) | C(16)]
    const float dA = __expf(dlt * a);
    h = dA * h + (dlt * u) * bc[n];
    float yv = h * bc[D_STATE + n];
    yv += __shfl_xor(yv, 1, 16);
    yv += __shfl_xor(yv, 2, 16);
    yv += __shfl_xor(yv, 4, 16);
    yv += __shfl_xor(yv, 8, 16);
    if (n == 0) {
      const float z  = xz[m * (2 * D_INNER) + D_INNER + d];
      const float sz = z / (1.f + __expf(-z));
      y[m * D_INNER + d] = (yv + u * Dd) * sz;
    }
  }
}

extern "C" void kernel_launch(void* const* d_in, const int* in_sizes, int n_in,
                              void* d_out, int out_size, void* d_ws, size_t ws_size,
                              hipStream_t stream)
{
  (void)in_sizes; (void)n_in; (void)out_size; (void)ws_size;
  const float* x        = (const float*)d_in[0];
  const float* w_in     = (const float*)d_in[1];
  const float* lA_in    = (const float*)d_in[2];
  const float* lB_in    = (const float*)d_in[3];
  const float* mask_in  = (const float*)d_in[4];
  const float* conv_w   = (const float*)d_in[5];
  const float* conv_b   = (const float*)d_in[6];
  const float* w_xproj  = (const float*)d_in[7];
  const float* w_dt     = (const float*)d_in[8];
  const float* b_dt     = (const float*)d_in[9];
  const float* A_log    = (const float*)d_in[10];
  const float* Dp       = (const float*)d_in[11];
  const float* w_out    = (const float*)d_in[12];
  const float* lA_out   = (const float*)d_in[13];
  const float* lB_out   = (const float*)d_in[14];
  const float* mask_out = (const float*)d_in[15];
  float* out = (float*)d_out;

  char* p = (char*)d_ws;
  auto alloc = [&](size_t bytes) -> char* {
    char* r = p; p += (bytes + 255) & ~(size_t)255; return r;
  };
  unsigned short* Abuf_in = (unsigned short*)alloc((size_t)MROWS * KP_IN * 2);
  unsigned short* Win_bf  = (unsigned short*)alloc((size_t)(2 * D_INNER) * KP_IN * 2);
  float*          xz      = (float*)alloc((size_t)MROWS * 2 * D_INNER * 4);
  float*          xs      = (float*)alloc((size_t)MROWS * D_INNER * 4);
  unsigned short* xs_bf   = (unsigned short*)alloc((size_t)MROWS * D_INNER * 2);
  unsigned short* Wxp_bf  = (unsigned short*)alloc((size_t)NXP * D_INNER * 2);
  float*          xdbl    = (float*)alloc((size_t)MROWS * NXP * 4);
  unsigned short* dt_bf   = (unsigned short*)alloc((size_t)MROWS * DT_RANK * 2);
  unsigned short* Wdt_bf  = (unsigned short*)alloc((size_t)D_INNER * DT_RANK * 2);
  float*          delta   = (float*)alloc((size_t)MROWS * D_INNER * 4);
  float*          yb      = (float*)alloc((size_t)MROWS * D_INNER * 4);
  unsigned short* Ybuf    = (unsigned short*)alloc((size_t)MROWS * KP_OUT * 2);
  unsigned short* Wout_bf = (unsigned short*)alloc((size_t)D_MODEL * KP_OUT * 2);

  // --- in_proj (base + LoRA folded into augmented K) ---
  pack_act<<<MROWS, 256, 0, stream>>>(x, lA_in, Abuf_in, D_MODEL, RANK, KP_IN);
  {
    size_t tot = (size_t)(2 * D_INNER) * KP_IN;
    pack_weight<<<(unsigned)((tot + 255) / 256), 256, 0, stream>>>(
        w_in, lB_in, mask_in, Win_bf, 2 * D_INNER, D_MODEL, RANK, KP_IN, LORA_SCALING);
  }
  {
    dim3 g((2 * D_INNER) / 128, MROWS / 128);
    gemm_bf16_wmma<<<g, 256, 0, stream>>>(Abuf_in, Win_bf, xz, nullptr,
                                          MROWS, 2 * D_INNER, KP_IN, 0);
  }
  // --- depthwise causal conv + SiLU ---
  conv_silu<<<(MROWS * D_INNER + 255) / 256, 256, 0, stream>>>(xz, conv_w, conv_b, xs, xs_bf);
  // --- x_proj ---
  {
    size_t tot = (size_t)NXP * D_INNER;
    pack_weight<<<(unsigned)((tot + 255) / 256), 256, 0, stream>>>(
        w_xproj, nullptr, nullptr, Wxp_bf, NXP, D_INNER, 0, D_INNER, 0.f);
    dim3 g((NXP + 127) / 128, MROWS / 128);
    gemm_bf16_wmma<<<g, 256, 0, stream>>>(xs_bf, Wxp_bf, xdbl, nullptr,
                                          MROWS, NXP, D_INNER, 0);
  }
  // --- dt_proj + softplus (fused epilogue) ---
  extract_dt<<<(MROWS * DT_RANK + 255) / 256, 256, 0, stream>>>(xdbl, dt_bf);
  {
    size_t tot = (size_t)D_INNER * DT_RANK;
    pack_weight<<<(unsigned)((tot + 255) / 256), 256, 0, stream>>>(
        w_dt, nullptr, nullptr, Wdt_bf, D_INNER, DT_RANK, 0, DT_RANK, 0.f);
    dim3 g(D_INNER / 128, MROWS / 128);
    gemm_bf16_wmma<<<g, 256, 0, stream>>>(dt_bf, Wdt_bf, delta, b_dt,
                                          MROWS, D_INNER, DT_RANK, 1);
  }
  // --- selective scan + SiLU(z) gate ---
  selective_scan<<<(BATCH * D_INNER * D_STATE) / 256, 256, 0, stream>>>(
      delta, xs, xdbl, A_log, Dp, xz, yb);
  // --- out_proj (base + LoRA folded) ---
  pack_act<<<MROWS, 256, 0, stream>>>(yb, lA_out, Ybuf, D_INNER, RANK, KP_OUT);
  {
    size_t tot = (size_t)D_MODEL * KP_OUT;
    pack_weight<<<(unsigned)((tot + 255) / 256), 256, 0, stream>>>(
        w_out, lB_out, mask_out, Wout_bf, D_MODEL, D_INNER, RANK, KP_OUT, LORA_SCALING);
    dim3 g(D_MODEL / 128, MROWS / 128);
    gemm_bf16_wmma<<<g, 256, 0, stream>>>(Ybuf, Wout_bf, out, nullptr,
                                          MROWS, D_MODEL, KP_OUT, 0);
  }
}
